// Bottleneck_DCNV3_11072425689154
// MI455X (gfx1250) — matrix operand order repaired and checked by reading, not compiled, not run on hardware
//
#include <hip/hip_runtime.h>
#include <hip/hip_bf16.h>
#include <math.h>

// ---------------- problem constants ----------------
#define NIMG 4
#define HH   64
#define WW   64
#define C1   256
#define CMID 128
#define C2   256
#define MTOT (NIMG*HH*WW)     // 16384 pixels
#define HP   66               // padded H (pad=1)
#define WP   66               // padded W
#define BN_EPS 1e-5f
#define LN_EPS 1e-6f

typedef __attribute__((ext_vector_type(16))) _Float16 v16h;
typedef __attribute__((ext_vector_type(8)))  _Float16 v8h;
typedef __attribute__((ext_vector_type(4)))  _Float16 v4h;
typedef __attribute__((ext_vector_type(8)))  float    v8f;
typedef __attribute__((ext_vector_type(4)))  int      v4i;

// ---- optional CDNA5 async global->LDS path (ASYNCcnt) ----
#ifdef __has_builtin
#if __has_builtin(__builtin_amdgcn_global_load_async_to_lds_b128) && \
    __has_builtin(__builtin_amdgcn_s_wait_asynccnt)
#define HAVE_ASYNC_LDS 1
#endif
#endif
#ifndef HAVE_ASYNC_LDS
#define HAVE_ASYNC_LDS 0
#endif

#if HAVE_ASYNC_LDS
// builtin signature (from hipcc diagnostic): first param is int4* in
// addrspace(1) ("__device__"); LDS side is addrspace(3).
typedef __attribute__((address_space(1))) v4i* gptr_v4i;
typedef __attribute__((address_space(3))) v4i* lptr_v4i;
#define ASYNC_CP16(dst_lds, src_glb)                                   \
    __builtin_amdgcn_global_load_async_to_lds_b128(                    \
        (gptr_v4i)(src_glb), (lptr_v4i)(dst_lds), 0, 0)
#endif

// ============================================================
// Packing kernels (fp32 -> f16, fragment-friendly layouts)
// ============================================================

// x (N,C1,H,W) f32 -> xpad (N,HP,WP,C1) f16 with zero ring
__global__ void k_pad_x(const float* __restrict__ x, _Float16* __restrict__ xpad) {
    int idx = blockIdx.x * blockDim.x + threadIdx.x;
    const int total = NIMG * HP * WP * C1;
    if (idx >= total) return;
    int c  = idx & 255;
    int t  = idx >> 8;
    int wp = t % WP; t /= WP;
    int hp = t % HP;
    int ni = t / HP;
    float v = 0.0f;
    if (hp >= 1 && hp <= HH && wp >= 1 && wp <= WW) {
        v = x[(((size_t)ni * C1 + c) * HH + (hp - 1)) * WW + (wp - 1)];
    }
    xpad[idx] = (_Float16)v;
}

// cv1_w (128,256,3,3) f32 -> B[(t*256+c)*128 + n] f16   (K=2304, N=128)
__global__ void k_pack_cv1w(const float* __restrict__ wsrc, _Float16* __restrict__ bdst) {
    int idx = blockIdx.x * blockDim.x + threadIdx.x;
    const int total = 9 * C1 * CMID;
    if (idx >= total) return;
    int n = idx % CMID;
    int c = (idx / CMID) % C1;
    int t = idx / (CMID * C1);
    bdst[idx] = (_Float16)wsrc[(size_t)n * (C1 * 9) + c * 9 + t];
}

// pw_w (256,128) f32 -> B[k*256 + n] f16   (K=128, N=256)
__global__ void k_pack_pw(const float* __restrict__ wsrc, _Float16* __restrict__ bdst) {
    int idx = blockIdx.x * blockDim.x + threadIdx.x;
    const int total = CMID * C2;
    if (idx >= total) return;
    int n = idx % C2;
    int k = idx / C2;
    bdst[idx] = (_Float16)wsrc[(size_t)n * CMID + k];
}

// W (256,256) f32 -> B[k*256 + n] f16  (transpose, K=256, N=256)
__global__ void k_pack_t256(const float* __restrict__ wsrc, _Float16* __restrict__ bdst) {
    int idx = blockIdx.x * blockDim.x + threadIdx.x;
    const int total = C2 * C2;
    if (idx >= total) return;
    int n = idx % C2;
    int k = idx / C2;
    bdst[idx] = (_Float16)wsrc[(size_t)n * C2 + k];
}

__global__ void k_zero32(unsigned int* __restrict__ p, int count) {
    int idx = blockIdx.x * blockDim.x + threadIdx.x;
    if (idx < count) p[idx] = 0u;
}

// ============================================================
// Implicit-conv WMMA GEMM, LDS-staged B panels, 16x64 per wave
//   C[m,n] = sum_k A_im2col[m,k] * B[k,n],  K = taps*Cin
//   block = 256 thr = 8 waves: 128 M-rows x 64 N-cols per block
//   B K-panel (32x64 f16 = 4KB) double-buffered in LDS;
//   staged with GLOBAL_LOAD_ASYNC_TO_LDS_B128 when available.
// store_mode: 0 = f16 rowmajor [M,Nout]
//             1 = f16 padded NHWC (HPxWP ring, Nout channels)
//             2 = f32 NCHW with residual add from xres -> out_f
// ============================================================
__global__ void __launch_bounds__(256)
wmma_conv_gemm(const _Float16* __restrict__ A,
               const _Float16* __restrict__ B,
               int Hp, int Wp, int cinShift, int taps, int Nout,
               const float* __restrict__ bias,
               const float* __restrict__ bn_g,
               const float* __restrict__ bn_b,
               int silu, int store_mode,
               _Float16* __restrict__ out_h,
               float* __restrict__ out_f,
               const float* __restrict__ xres) {
    const int tid  = threadIdx.x;
    const int lane = tid & 31;
    const int wv   = tid >> 5;

    const int Cin     = 1 << cinShift;
    const int Ngroups = Nout >> 6;
    const int sm  = blockIdx.x / Ngroups;   // M-supertile (128 rows)
    const int ng  = blockIdx.x % Ngroups;   // 64-col N group
    const int m0  = sm * 128 + wv * 16;
    const int n0g = ng * 64;

    // per-lane A row: M = lane&15
    const int m_row = m0 + (lane & 15);
    const int nimg  = m_row >> 12;
    const int rem   = m_row & 4095;
    const int h     = rem >> 6;
    const int w     = rem & 63;
    const int khalf = (lane >> 4) << 3;     // +8 halves for lanes 16..31

    __shared__ _Float16 bpanel[2][32 * 64]; // two 4KB K-panels

    const int nk = (taps << cinShift) >> 5; // K / 32

    // cooperative panel loader: thread tid moves 16B
    const int prow  = tid >> 3;             // 0..31  (k within chunk)
    const int pcol  = (tid & 7) << 3;       // 0..56  (halves)
    const _Float16* gpan0 = B + (size_t)prow * Nout + n0g + pcol;
    const size_t    gstep = (size_t)32 * Nout;   // halves per K-chunk

    // ---- preload chunk 0 ----
    {
        _Float16* dst = &bpanel[0][prow * 64 + pcol];
#if HAVE_ASYNC_LDS
        ASYNC_CP16(dst, gpan0);
        __builtin_amdgcn_s_wait_asynccnt(0);
#else
        *(v8h*)dst = *(const v8h*)gpan0;
#endif
    }
    __syncthreads();

    v8f acc[4];
#pragma unroll
    for (int j = 0; j < 4; ++j)
        acc[j] = (v8f){0.f, 0.f, 0.f, 0.f, 0.f, 0.f, 0.f, 0.f};

    for (int kc = 0; kc < nk; ++kc) {
        const int cur = kc & 1;

        // stage next K-panel into the other LDS buffer
        if (kc + 1 < nk) {
            const _Float16* gsrc = gpan0 + (size_t)(kc + 1) * gstep;
            _Float16* dst = &bpanel[1 - cur][prow * 64 + pcol];
#if HAVE_ASYNC_LDS
            ASYNC_CP16(dst, gsrc);
#else
            *(v8h*)dst = *(const v8h*)gsrc;
#endif
        }

        // A fragment for this K-chunk (from global, per-wave rows)
        const int k0  = kc << 5;
        const int t   = k0 >> cinShift;
        int ty = 0, tx = 0;
        if (taps == 9) { ty = t / 3; tx = t - ty * 3; }
        const int cb = (k0 & (Cin - 1)) + khalf;
        const _Float16* ap =
            A + ((size_t)((nimg * Hp + h + ty) * Wp) + (size_t)(w + tx)) * (size_t)Cin + cb;
        union { v16h v; v8h h8[2]; } af;
        af.h8[0] = *(const v8h*)(ap);
        af.h8[1] = *(const v8h*)(ap + 16);
        __builtin_prefetch(ap + 32, 0, 1);      // next chunk of this row

        // 4 sub-tiles of B from LDS, A fragment reused 4x
        const _Float16* brow = &bpanel[cur][((lane >> 4) << 3) * 64 + (lane & 15)];
#pragma unroll
        for (int j = 0; j < 4; ++j) {
            union { v16h v; v8h h8[2]; } bf;
#pragma unroll
            for (int s = 0; s < 8; ++s) {
                bf.v[s]     = brow[s * 64 + j * 16];
                bf.v[8 + s] = brow[(16 + s) * 64 + j * 16];
            }
            acc[j] = __builtin_amdgcn_wmma_f32_16x16x32_f16(
                false, af.v, false, bf.v, (short)0, acc[j], false, false);
        }

#if HAVE_ASYNC_LDS
        __builtin_amdgcn_s_wait_asynccnt(0);
#endif
        __syncthreads();
    }

    // epilogue: lane n = base + (lane&15); vgpr r -> m = m0 + r + 8*(lane>=16)
    const int mbase = m0 + ((lane >> 4) << 3);
#pragma unroll
    for (int j = 0; j < 4; ++j) {
        const int n = n0g + j * 16 + (lane & 15);
        const float bs = bias ? bias[n] : 0.0f;
        float sc = 1.0f, sb = 0.0f;
        if (bn_g) { sc = bn_g[n] * rsqrtf(1.0f + BN_EPS); sb = bn_b[n]; }
#pragma unroll
        for (int r = 0; r < 8; ++r) {
            float v = acc[j][r] + bs;
            if (bn_g) v = v * sc + sb;
            if (silu) v = v / (1.0f + __expf(-v));
            const int m = mbase + r;
            if (store_mode == 0) {
                out_h[(size_t)m * Nout + n] = (_Float16)v;
            } else if (store_mode == 1) {
                const int ni = m >> 12, rr = m & 4095, hh = rr >> 6, ww2 = rr & 63;
                out_h[((size_t)((ni * HP + hh + 1) * WP) + (size_t)(ww2 + 1)) * (size_t)Nout + n] =
                    (_Float16)v;
            } else {
                const int ni = m >> 12, rr = m & 4095, hh = rr >> 6, ww2 = rr & 63;
                const size_t idx = (((size_t)ni * C2 + n) * HH + hh) * WW + ww2;
                out_f[idx] = xres[idx] + v;
            }
        }
    }
}

// ============================================================
// depthwise 3x3 + bias + LayerNorm(C) + GELU(exact)
// one wave32 per pixel, 8 channels per lane
// ============================================================
__global__ void k_dw_ln_gelu(const _Float16* __restrict__ y2,
                             const float* __restrict__ dw_w,
                             const float* __restrict__ dw_b,
                             const float* __restrict__ ln_g,
                             const float* __restrict__ ln_b,
                             _Float16* __restrict__ x1) {
    const int m = blockIdx.x;
    const int lane = threadIdx.x;
    const int nimg = m >> 12;
    const int h = (m >> 6) & 63;
    const int w = m & 63;
    const int c0 = lane * 8;

    float acc[8];
#pragma unroll
    for (int j = 0; j < 8; ++j) acc[j] = 0.0f;

    for (int t = 0; t < 9; ++t) {
        const int hh = h + t / 3 - 1;
        const int ww2 = w + (t % 3) - 1;
        if (hh < 0 || hh >= HH || ww2 < 0 || ww2 >= WW) continue;
        const size_t row = (size_t)(nimg * 4096 + hh * 64 + ww2) * C2 + c0;
        v8h v = *(const v8h*)(y2 + row);
#pragma unroll
        for (int j = 0; j < 8; ++j)
            acc[j] += (float)v[j] * dw_w[(size_t)(c0 + j) * 9 + t];
    }

    float vals[8];
    float s = 0.0f, s2 = 0.0f;
#pragma unroll
    for (int j = 0; j < 8; ++j) {
        vals[j] = acc[j] + dw_b[c0 + j];
        s += vals[j];
        s2 += vals[j] * vals[j];
    }
#pragma unroll
    for (int o = 16; o > 0; o >>= 1) {
        s  += __shfl_xor(s,  o, 32);
        s2 += __shfl_xor(s2, o, 32);
    }
    const float mean = s * (1.0f / C2);
    const float var  = s2 * (1.0f / C2) - mean * mean;
    const float rstd = rsqrtf(var + LN_EPS);
#pragma unroll
    for (int j = 0; j < 8; ++j) {
        float y = (vals[j] - mean) * rstd * ln_g[c0 + j] + ln_b[c0 + j];
        float g = 0.5f * y * (1.0f + erff(y * 0.70710678118f));
        x1[(size_t)m * C2 + c0 + j] = (_Float16)g;
    }
}

// ============================================================
// offset/mask linears (27 outputs from 256) + softmax over 9
// one wave32 per pixel
// ============================================================
__global__ void k_offmask(const _Float16* __restrict__ x1,
                          const float* __restrict__ off_w, const float* __restrict__ off_b,
                          const float* __restrict__ msk_w, const float* __restrict__ msk_b,
                          float* __restrict__ off, float* __restrict__ msk) {
    __shared__ _Float16 row[C2];
    __shared__ float vals[32];
    const int m = blockIdx.x;
    const int lane = threadIdx.x;

    *(v8h*)&row[lane * 8] = *(const v8h*)(x1 + (size_t)m * C2 + lane * 8);
    __syncthreads();

    if (lane < 27) {
        const float* Wr = (lane < 18) ? (off_w + (size_t)lane * C2)
                                      : (msk_w + (size_t)(lane - 18) * C2);
        float a = (lane < 18) ? off_b[lane] : msk_b[lane - 18];
        for (int k = 0; k < C2; ++k) a += (float)row[k] * Wr[k];
        vals[lane] = a;
    }
    __syncthreads();

    if (lane < 18) {
        off[(size_t)m * 18 + lane] = vals[lane];
    } else if (lane < 27) {
        float mx = -1e30f;
        for (int j = 18; j < 27; ++j) mx = fmaxf(mx, vals[j]);
        float ssum = 0.0f;
        for (int j = 18; j < 27; ++j) ssum += __expf(vals[j] - mx);
        msk[(size_t)m * 9 + (lane - 18)] = __expf(vals[lane] - mx) / ssum;
    }
}

// ============================================================
// DCNv3 core: 9-tap bilinear gather over padded f16 xproj
// 64 threads per pixel, 4 channels per thread
// ============================================================
__global__ void k_dcnv3(const _Float16* __restrict__ xprojpad,
                        const float* __restrict__ off,
                        const float* __restrict__ msk,
                        _Float16* __restrict__ core) {
    __shared__ float spx[9], spy[9], smk[9];
    const int m = blockIdx.x;
    const int tid = threadIdx.x;
    const int nimg = m >> 12;
    const int h = (m >> 6) & 63;
    const int w = m & 63;

    if (tid < 9) {
        const int p = tid;
        const float gx = (float)(p / 3 - 1);   // meshgrid 'ij': x varies with p/3
        const float gy = (float)(p % 3 - 1);
        const float ox = off[(size_t)m * 18 + 2 * p];
        const float oy = off[(size_t)m * 18 + 2 * p + 1];
        spx[p] = (float)w + 1.0f + gx + ox;    // loc_x*WP - 0.5 expanded
        spy[p] = (float)h + 1.0f + gy + oy;
        smk[p] = msk[(size_t)m * 9 + p];
    }
    __syncthreads();

    const int c0 = tid * 4;
    float a0 = 0.f, a1 = 0.f, a2 = 0.f, a3 = 0.f;
    const _Float16* img = xprojpad + (size_t)nimg * HP * WP * C2;

    for (int p = 0; p < 9; ++p) {
        const float px = spx[p], py = spy[p], mm = smk[p];
        const float x0f = floorf(px), y0f = floorf(py);
        const float fx = px - x0f, fy = py - y0f;
        const int ix = (int)x0f, iy = (int)y0f;
#pragma unroll
        for (int dy = 0; dy < 2; ++dy) {
#pragma unroll
            for (int dx = 0; dx < 2; ++dx) {
                const int xi = ix + dx, yi = iy + dy;
                if (xi < 0 || xi >= WP || yi < 0 || yi >= HP) continue;
                const float wt = (dx ? fx : 1.0f - fx) * (dy ? fy : 1.0f - fy) * mm;
                v4h v = *(const v4h*)(img + ((size_t)yi * WP + xi) * C2 + c0);
                a0 += wt * (float)v[0];
                a1 += wt * (float)v[1];
                a2 += wt * (float)v[2];
                a3 += wt * (float)v[3];
            }
        }
    }
    _Float16* dst = core + (size_t)m * C2 + c0;
    dst[0] = (_Float16)a0; dst[1] = (_Float16)a1;
    dst[2] = (_Float16)a2; dst[3] = (_Float16)a3;
}

// ============================================================
// host-side launch
// ============================================================
extern "C" void kernel_launch(void* const* d_in, const int* in_sizes, int n_in,
                              void* d_out, int out_size, void* d_ws, size_t ws_size,
                              hipStream_t stream) {
    const float* x      = (const float*)d_in[0];
    const float* cv1_w  = (const float*)d_in[1];
    const float* cv1_g  = (const float*)d_in[2];
    const float* cv1_b  = (const float*)d_in[3];
    const float* pw_w   = (const float*)d_in[4];
    const float* pw_g   = (const float*)d_in[5];
    const float* pw_b   = (const float*)d_in[6];
    const float* dw_w   = (const float*)d_in[7];
    const float* dw_b   = (const float*)d_in[8];
    const float* ln_g   = (const float*)d_in[9];
    const float* ln_b   = (const float*)d_in[10];
    const float* off_w  = (const float*)d_in[11];
    const float* off_b  = (const float*)d_in[12];
    const float* msk_w  = (const float*)d_in[13];
    const float* msk_b  = (const float*)d_in[14];
    const float* inp_w  = (const float*)d_in[15];
    const float* inp_b  = (const float*)d_in[16];
    const float* out_w  = (const float*)d_in[17];
    const float* out_b  = (const float*)d_in[18];
    const float* bn2_g  = (const float*)d_in[19];
    const float* bn2_b  = (const float*)d_in[20];
    float* out_f = (float*)d_out;

    // carve workspace (256B aligned slices)
    size_t cur = 0;
    auto carve = [&](size_t bytes) -> void* {
        void* p = (char*)d_ws + cur;
        cur += (bytes + 255) & ~(size_t)255;
        return p;
    };
    _Float16* xpad     = (_Float16*)carve((size_t)NIMG * HP * WP * C1 * 2);
    _Float16* Bc1      = (_Float16*)carve((size_t)9 * C1 * CMID * 2);
    _Float16* y1       = (_Float16*)carve((size_t)MTOT * CMID * 2);
    _Float16* Bpw      = (_Float16*)carve((size_t)CMID * C2 * 2);
    _Float16* y2       = (_Float16*)carve((size_t)MTOT * C2 * 2);
    _Float16* Binp     = (_Float16*)carve((size_t)C2 * C2 * 2);
    _Float16* xprojpad = (_Float16*)carve((size_t)NIMG * HP * WP * C2 * 2);
    _Float16* x1       = (_Float16*)carve((size_t)MTOT * C2 * 2);
    float*    offb     = (float*)carve((size_t)MTOT * 18 * 4);
    float*    mskb     = (float*)carve((size_t)MTOT * 9 * 4);
    _Float16* coreb    = (_Float16*)carve((size_t)MTOT * C2 * 2);
    _Float16* Bout     = (_Float16*)carve((size_t)C2 * C2 * 2);

    // ---- packing ----
    {
        int n = NIMG * HP * WP * C1;
        k_pad_x<<<(n + 255) / 256, 256, 0, stream>>>(x, xpad);
    }
    {
        int n = 9 * C1 * CMID;
        k_pack_cv1w<<<(n + 255) / 256, 256, 0, stream>>>(cv1_w, Bc1);
    }
    {
        int n = CMID * C2;
        k_pack_pw<<<(n + 255) / 256, 256, 0, stream>>>(pw_w, Bpw);
    }
    {
        int n = C2 * C2;
        k_pack_t256<<<(n + 255) / 256, 256, 0, stream>>>(inp_w, Binp);
        k_pack_t256<<<(n + 255) / 256, 256, 0, stream>>>(out_w, Bout);
    }
    {
        int n = NIMG * HP * WP * C2 / 2;  // u32 count
        k_zero32<<<(n + 255) / 256, 256, 0, stream>>>((unsigned int*)xprojpad, n);
    }

    const int Msupers = MTOT / 128;   // 128 blocks of 8 waves x 16 rows

    // ---- cv1: 3x3 conv, K=2304 (Cin=256 shift 8), N=128, BN+SiLU -> y1 ----
    wmma_conv_gemm<<<Msupers * (CMID / 64), 256, 0, stream>>>(
        xpad, Bc1, HP, WP, 8, 9, CMID,
        nullptr, cv1_g, cv1_b, 1, 0, y1, nullptr, nullptr);

    // ---- pw: 1x1 conv, K=128 (shift 7), N=256, BN+SiLU -> y2 ----
    wmma_conv_gemm<<<Msupers * (C2 / 64), 256, 0, stream>>>(
        y1, Bpw, HH, WW, 7, 1, C2,
        nullptr, pw_g, pw_b, 1, 0, y2, nullptr, nullptr);

    // ---- input_proj: K=256 (shift 8), N=256, +bias -> padded NHWC xproj ----
    wmma_conv_gemm<<<Msupers * (C2 / 64), 256, 0, stream>>>(
        y2, Binp, HH, WW, 8, 1, C2,
        inp_b, nullptr, nullptr, 0, 1, xprojpad, nullptr, nullptr);

    // ---- depthwise 3x3 + LN + GELU -> x1 f16 ----
    k_dw_ln_gelu<<<MTOT, 32, 0, stream>>>(y2, dw_w, dw_b, ln_g, ln_b, x1);

    // ---- offset / mask + softmax ----
    k_offmask<<<MTOT, 32, 0, stream>>>(x1, off_w, off_b, msk_w, msk_b, offb, mskb);

    // ---- DCNv3 bilinear gather core -> core f16 ----
    k_dcnv3<<<MTOT, 64, 0, stream>>>(xprojpad, offb, mskb, coreb);

    // ---- output_proj + bias + BN2 + SiLU + residual -> d_out (f32 NCHW) ----
    wmma_conv_gemm<<<Msupers * (C2 / 64), 256, 0, stream>>>(
        coreb, Bout, HH, WW, 8, 1, C2,
        out_b, bn2_g, bn2_b, 1, 2, nullptr, out_f, x);
}
